// CrossAttention_6820408066620
// MI455X (gfx1250) — matrix operand order repaired and checked by reading, not compiled, not run on hardware
//
#include <hip/hip_runtime.h>
#include <hip/hip_bf16.h>

// Problem constants (from reference): B=8, H=W=64 -> N=4096, d=64, in_ch=19
#define BATCH 8
#define NPIX  4096
#define DHEAD 64
#define INCH  19
#define TK    64      // keys per main-loop step
#define NWAVE 8       // waves per block
#define KROW  72      // padded LDS row (elements) for 64-wide tiles: 144B = 36 banks
#define SROW  40      // padded LDS row (elements) for 32-wide tiles:  80B = 20 banks

typedef __attribute__((ext_vector_type(16))) __bf16 v16bf;
typedef __attribute__((ext_vector_type(8)))  __bf16 v8bf;
typedef __attribute__((ext_vector_type(8)))  float  v8f;

union BF16x16 { v16bf v; v8bf h[2]; };

__device__ __forceinline__ v8f wmma_bf16(v16bf a, v16bf b, v8f c) {
  // D = A(16x32 bf16) * B(32x16 bf16) + C(16x16 f32)
  return __builtin_amdgcn_wmma_f32_16x16x32_bf16(false, a, false, b, (short)0, c,
                                                 false, false);
}

// ---- DPP16 XOR-style reductions within each 16-lane row (pure VALU, no LDS).
template <int CTRL>
__device__ __forceinline__ float dpp_f(float x) {
  return __builtin_bit_cast(float, __builtin_amdgcn_update_dpp(
      0, __builtin_bit_cast(int, x), CTRL, 0xF, 0xF, true));
}
__device__ __forceinline__ float redmax16(float x) {
  x = fmaxf(x, dpp_f<0xB1>(x));    // quad_perm [1,0,3,2] : xor 1
  x = fmaxf(x, dpp_f<0x4E>(x));    // quad_perm [2,3,0,1] : xor 2
  x = fmaxf(x, dpp_f<0x141>(x));   // row_half_mirror     : xor 4
  x = fmaxf(x, dpp_f<0x140>(x));   // row_mirror          : xor 8
  return x;
}
__device__ __forceinline__ float redsum16(float x) {
  x += dpp_f<0xB1>(x);
  x += dpp_f<0x4E>(x);
  x += dpp_f<0x141>(x);
  x += dpp_f<0x140>(x);
  return x;
}

// ---- CDNA5 async global->LDS copy (ASYNCcnt), 16B per lane.
__device__ __forceinline__ void async_b128(unsigned lds_off, const void* gaddr) {
  asm volatile("global_load_async_to_lds_b128 %0, %1, off"
               :: "v"(lds_off), "v"(gaddr) : "memory");
}

// ---------------------------------------------------------------------------
// Kernel 1: per-pixel projections; the 19->64 Q projection is a tall-skinny
// GEMM and runs on WMMA (K padded 19->32 with zeros).
//   Q[b][n][c]  = Wq[c,:] . seg[b,:,n] + bq[c]      (bf16, pixel-major)
//   K[b][n][c]  = Wk[c]*d[b,n] + bk[c]              (bf16, pixel-major)
//   Vt[b][c][n] = Wv[c]*d[b,n] + bv[c]              (bf16, channel-major)
// ---------------------------------------------------------------------------
__global__ __launch_bounds__(256)
void qkv_prep(const float* __restrict__ seg, const float* __restrict__ dep,
              const float* __restrict__ Wq,  const float* __restrict__ bq,
              const float* __restrict__ Wk,  const float* __restrict__ bk,
              const float* __restrict__ Wv,  const float* __restrict__ bv,
              __bf16* __restrict__ Q, __bf16* __restrict__ Kd,
              __bf16* __restrict__ Vt)
{
  __shared__ __attribute__((aligned(16))) __bf16 segT[256][SROW]; // pixel-major, c padded
  __shared__ __attribute__((aligned(16))) __bf16 Wqb[64][SROW];   // o-major Wq^T, c padded
  __shared__ float sWk[DHEAD], sbk[DHEAD], sWv[DHEAD], sbv[DHEAD], sbq[DHEAD];

  const int tid = threadIdx.x;
  const int b   = blockIdx.x >> 4;          // 16 blocks per batch
  const int n0  = (blockIdx.x & 15) * 256;

  // Stage Wq^T as bf16 (zero-padded K) + scalar params.
  for (int i = tid; i < 64 * 32; i += 256) {
    const int o = i >> 5, c = i & 31;
    Wqb[o][c] = (c < INCH) ? (__bf16)Wq[o * INCH + c] : (__bf16)0.0f;
  }
  if (tid < DHEAD) {
    sWk[tid] = Wk[tid]; sbk[tid] = bk[tid];
    sWv[tid] = Wv[tid]; sbv[tid] = bv[tid]; sbq[tid] = bq[tid];
  }
  // Stage seg tile transposed to pixel-major bf16 (coalesced global reads).
  {
    const float* sp = seg + (size_t)b * INCH * NPIX + n0 + tid;
#pragma unroll
    for (int c = 0; c < INCH; ++c) segT[tid][c] = (__bf16)sp[(size_t)c * NPIX];
#pragma unroll
    for (int c = INCH; c < 32; ++c) segT[tid][c] = (__bf16)0.0f;
  }
  __syncthreads();

  const int lane = tid & 31, wave = tid >> 5;
  const int hi = lane >> 4, c16 = lane & 15;

  // WMMA Q projection: each wave does two 16-pixel tiles x 4 o-tiles.
#pragma unroll
  for (int t = 0; t < 2; ++t) {
    const int ptile = wave * 2 + t;          // 0..15
    const __bf16* arow = &segT[ptile * 16 + c16][0] + hi * 8;
    BF16x16 a;
    a.h[0] = *(const v8bf*)(arow);           // k {0..7}/{8..15}
    a.h[1] = *(const v8bf*)(arow + 16);      // k {16..23}/{24..31}
#pragma unroll
    for (int ot = 0; ot < 4; ++ot) {
      const int o = ot * 16 + c16;
      const __bf16* brow = &Wqb[o][0] + hi * 8;
      BF16x16 bb;
      bb.h[0] = *(const v8bf*)(brow);
      bb.h[1] = *(const v8bf*)(brow + 16);
      v8f d;
#pragma unroll
      for (int r = 0; r < 8; ++r) d[r] = 0.0f;
      d = wmma_bf16(a.v, bb.v, d);
      const float bqo = sbq[o];
      __bf16* qbase = Q + ((size_t)b * NPIX + n0 + ptile * 16 + 8 * hi) * DHEAD + o;
#pragma unroll
      for (int r = 0; r < 8; ++r) qbase[(size_t)r * DHEAD] = (__bf16)(d[r] + bqo);
    }
  }

  // K and Vt are rank-1 + bias: per-thread scalar.
  const float dv = dep[(size_t)b * NPIX + n0 + tid];
  __bf16* Krow = Kd + ((size_t)b * NPIX + n0 + tid) * DHEAD;
#pragma unroll
  for (int o8 = 0; o8 < 8; ++o8) {
    v8bf kp;
#pragma unroll
    for (int j = 0; j < 8; ++j) {
      const int o = o8 * 8 + j;
      kp[j] = (__bf16)(sWk[o] * dv + sbk[o]);
      Vt[((size_t)b * DHEAD + o) * NPIX + n0 + tid] = (__bf16)(sWv[o] * dv + sbv[o]);
    }
    *(v8bf*)(Krow + o8 * 8) = kp;
  }
}

// Stage one 64-key K tile ([key][c], padded rows) and V tile ([c][key], padded
// rows) into LDS: 2 x 512 x 16B chunks, 256 threads x 2 chunks each.
__device__ __forceinline__ void stage_tiles(const __bf16* Kb, const __bf16* Vb,
                                            int i0, unsigned klds, unsigned vlds,
                                            int tid) {
#pragma unroll
  for (int j = 0; j < 2; ++j) {
    const int chunk = tid + 256 * j;
    const int row = chunk >> 3, off = chunk & 7;
    async_b128(klds + row * (KROW * 2) + off * 16,
               (const char*)(Kb + (size_t)(i0 + row) * DHEAD) + (size_t)off * 16);
    async_b128(vlds + row * (KROW * 2) + off * 16,
               (const char*)(Vb + (size_t)row * NPIX + i0) + (size_t)off * 16);
  }
}

// ---------------------------------------------------------------------------
// Kernel 2: fused flash attention + output projection on WMMA.
// One wave = 16 query rows; block = 8 waves = 128 rows. 64 keys per step,
// K/V tiles double-buffered in LDS via global_load_async_to_lds_b128.
// ---------------------------------------------------------------------------
__global__ __launch_bounds__(256)
void fa_kernel(const __bf16* __restrict__ Q, const __bf16* __restrict__ K,
               const __bf16* __restrict__ Vt, const float* __restrict__ Wo,
               const float* __restrict__ bo, float* __restrict__ out)
{
  __shared__ __attribute__((aligned(16))) __bf16 ktile[2][TK * KROW];    // [key][c]
  __shared__ __attribute__((aligned(16))) __bf16 vtile[2][DHEAD * KROW]; // [c][key]
  __shared__ __attribute__((aligned(16))) __bf16 pbuf[NWAVE][16 * KROW];

  const int tid  = threadIdx.x;
  const int lane = tid & 31;
  const int wave = tid >> 5;
  const int hi   = lane >> 4;
  const int c16  = lane & 15;
  const int b    = blockIdx.x >> 5;                       // 32 blocks per batch
  const int mbase = ((blockIdx.x & 31) * NWAVE + wave) * 16;

  const __bf16* Kb = K  + (size_t)b * NPIX * DHEAD;
  const __bf16* Vb = Vt + (size_t)b * DHEAD * NPIX;
  __bf16* Pl = &pbuf[wave][0];

  // Q A-operands (A 16x32: lane<16 holds K in {0..7,16..23}, lane>=16 {8..15,24..31})
  const __bf16* qrow = Q + ((size_t)b * NPIX + mbase + c16) * DHEAD + hi * 8;
  BF16x16 qa0, qa1;
  qa0.h[0] = *(const v8bf*)(qrow);       qa0.h[1] = *(const v8bf*)(qrow + 16);
  qa1.h[0] = *(const v8bf*)(qrow + 32);  qa1.h[1] = *(const v8bf*)(qrow + 48);

  v8f oacc[4];
  float rmax[8], rsum[8];
#pragma unroll
  for (int t = 0; t < 4; ++t)
#pragma unroll
    for (int r = 0; r < 8; ++r) oacc[t][r] = 0.0f;
#pragma unroll
  for (int r = 0; r < 8; ++r) { rmax[r] = -3.0e38f; rsum[r] = 0.0f; }

  const unsigned klds[2] = { (unsigned)(uintptr_t)&ktile[0][0],
                             (unsigned)(uintptr_t)&ktile[1][0] };
  const unsigned vlds[2] = { (unsigned)(uintptr_t)&vtile[0][0],
                             (unsigned)(uintptr_t)&vtile[1][0] };

  stage_tiles(Kb, Vb, 0, klds[0], vlds[0], tid);

  int buf = 0;
  for (int i0 = 0; i0 < NPIX; i0 += TK, buf ^= 1) {
    asm volatile("s_wait_asynccnt 0" ::: "memory");  // our tile-i loads done
    __syncthreads();                                  // everyone's loads done,
                                                      // everyone done reading other buf
    if (i0 + TK < NPIX)
      stage_tiles(Kb, Vb, i0 + TK, klds[buf ^ 1], vlds[buf ^ 1], tid);

    const __bf16* kt = &ktile[buf][0];
    const __bf16* vt = &vtile[buf][0];

    // ---- S = Q*K^T for 16 x 64 keys : 4 tiles x 2 c-windows = 8 WMMAs ------
    v8f s[4];
#pragma unroll
    for (int it = 0; it < 4; ++it) {
#pragma unroll
      for (int r = 0; r < 8; ++r) s[it][r] = 0.0f;
      const __bf16* krow = kt + (size_t)(it * 16 + c16) * KROW + hi * 8;
      BF16x16 kb;
      kb.h[0] = *(const v8bf*)(krow);      kb.h[1] = *(const v8bf*)(krow + 16);
      s[it] = wmma_bf16(qa0.v, kb.v, s[it]);
      kb.h[0] = *(const v8bf*)(krow + 32); kb.h[1] = *(const v8bf*)(krow + 48);
      s[it] = wmma_bf16(qa1.v, kb.v, s[it]);
    }

    // ---- online softmax (DPP16 row reductions; C/D layout: lane=col,
    //      VGPR r = row r + 8*hi) -------------------------------------------
#pragma unroll
    for (int r = 0; r < 8; ++r) {
      float tm = fmaxf(fmaxf(s[0][r], s[1][r]), fmaxf(s[2][r], s[3][r]));
      tm = redmax16(tm);
      const float mnew = fmaxf(rmax[r], tm);
      const float corr = __expf(rmax[r] - mnew);
      rmax[r] = mnew;
      float p[4];
#pragma unroll
      for (int it = 0; it < 4; ++it) p[it] = __expf(s[it][r] - mnew);
      float ps = (p[0] + p[1]) + (p[2] + p[3]);
      ps = redsum16(ps);
      rsum[r] = rsum[r] * corr + ps;
#pragma unroll
      for (int t = 0; t < 4; ++t) oacc[t][r] *= corr;
#pragma unroll
      for (int it = 0; it < 4; ++it)
        Pl[(r + 8 * hi) * KROW + it * 16 + c16] = (__bf16)p[it];
    }
    asm volatile("s_wait_dscnt 0" ::: "memory");     // intra-wave P store->load

    // Reload P as A-operands for the two 32-key windows.
    BF16x16 pa0, pa1;
    const __bf16* prow = Pl + c16 * KROW + hi * 8;
    pa0.h[0] = *(const v8bf*)(prow);      pa0.h[1] = *(const v8bf*)(prow + 16);
    pa1.h[0] = *(const v8bf*)(prow + 32); pa1.h[1] = *(const v8bf*)(prow + 48);

    // ---- O += P*V : 4 channel tiles x 2 key windows = 8 WMMAs --------------
#pragma unroll
    for (int ot = 0; ot < 4; ++ot) {
      const __bf16* vrow = vt + (size_t)(ot * 16 + c16) * KROW + hi * 8;
      BF16x16 vb;
      vb.h[0] = *(const v8bf*)(vrow);      vb.h[1] = *(const v8bf*)(vrow + 16);
      oacc[ot] = wmma_bf16(pa0.v, vb.v, oacc[ot]);
      vb.h[0] = *(const v8bf*)(vrow + 32); vb.h[1] = *(const v8bf*)(vrow + 48);
      oacc[ot] = wmma_bf16(pa1.v, vb.v, oacc[ot]);
    }
  }

  // ---- normalize by row sums ----------------------------------------------
#pragma unroll
  for (int r = 0; r < 8; ++r) {
    const float inv = 1.0f / rsum[r];
#pragma unroll
    for (int t = 0; t < 4; ++t) oacc[t][r] *= inv;
  }

  // ---- fused output projection: D[m][o] = sum_c O[m][c]*Wo[o][c] + bo[o] ---
#pragma unroll
  for (int t = 0; t < 4; ++t)
#pragma unroll
    for (int r = 0; r < 8; ++r)
      Pl[(r + 8 * hi) * KROW + t * 16 + c16] = (__bf16)oacc[t][r];
  asm volatile("s_wait_dscnt 0" ::: "memory");

  BF16x16 a0, a1;
  const __bf16* orow = Pl + c16 * KROW + hi * 8;
  a0.h[0] = *(const v8bf*)orow;        a0.h[1] = *(const v8bf*)(orow + 16);
  a1.h[0] = *(const v8bf*)(orow + 32); a1.h[1] = *(const v8bf*)(orow + 48);

#pragma unroll
  for (int ot = 0; ot < 4; ++ot) {
    const int oc = ot * 16 + c16;
    const float* wrow = Wo + oc * DHEAD + hi * 8;
    BF16x16 w0, w1;
#pragma unroll
    for (int j = 0; j < 8; ++j) {
      w0.h[0][j] = (__bf16)wrow[j];       w0.h[1][j] = (__bf16)wrow[16 + j];
      w1.h[0][j] = (__bf16)wrow[32 + j];  w1.h[1][j] = (__bf16)wrow[48 + j];
    }
    v8f d;
#pragma unroll
    for (int r = 0; r < 8; ++r) d[r] = 0.0f;
    d = wmma_bf16(a0.v, w0.v, d);
    d = wmma_bf16(a1.v, w1.v, d);
    const float bov = bo[oc];
    float* obase = out + ((size_t)b * DHEAD + oc) * NPIX + mbase + 8 * hi;
#pragma unroll
    for (int r = 0; r < 8; ++r) obase[r] = d[r] + bov;
  }
}

// ---------------------------------------------------------------------------
extern "C" void kernel_launch(void* const* d_in, const int* in_sizes, int n_in,
                              void* d_out, int out_size, void* d_ws, size_t ws_size,
                              hipStream_t stream) {
  (void)in_sizes; (void)n_in; (void)out_size; (void)ws_size;
  const float* seg = (const float*)d_in[0];
  const float* dep = (const float*)d_in[1];
  const float* Wq  = (const float*)d_in[2];
  const float* bq  = (const float*)d_in[3];
  const float* Wk  = (const float*)d_in[4];
  const float* bk  = (const float*)d_in[5];
  const float* Wv  = (const float*)d_in[6];
  const float* bv  = (const float*)d_in[7];
  const float* Wo  = (const float*)d_in[8];
  const float* bo  = (const float*)d_in[9];

  // Workspace (12 MiB): Q | K (pixel-major bf16), Vt (channel-major bf16)
  __bf16* Q  = (__bf16*)d_ws;
  __bf16* K  = Q + (size_t)BATCH * NPIX * DHEAD;
  __bf16* Vt = K + (size_t)BATCH * NPIX * DHEAD;

  qkv_prep<<<BATCH * (NPIX / 256), 256, 0, stream>>>(seg, dep, Wq, bq, Wk, bk,
                                                     Wv, bv, Q, K, Vt);
  fa_kernel<<<BATCH * (NPIX / 128), 256, 0, stream>>>(Q, K, Vt, Wo, bo,
                                                      (float*)d_out);
}
